// GPTJAttention_29678224015360
// MI455X (gfx1250) — matrix operand order repaired and checked by reading, not compiled, not run on hardware
//
#include <hip/hip_runtime.h>

typedef __bf16 bf16;
typedef __attribute__((ext_vector_type(16))) __bf16 v16bf;
typedef __attribute__((ext_vector_type(8)))  __bf16 bf16x8;
typedef __attribute__((ext_vector_type(8)))  float   v8f;

#define B_   4
#define S_   2048
#define H_   4096
#define NH_  16
#define HD_  256
#define N3_  12288
#define RD_  64
#define SCALE_ 0.0625f          // HD^-0.5
#define NEG_LOG_THETA_OVER_RD 0.1439115682f   // ln(10000)/64

// ---------------------------------------------------------------- WMMA helpers
__device__ __forceinline__ v8f wmma_bf16(v16bf a, v16bf b, v8f c) {
  // D = A(16x32) * B(32x16) + C, f32 accumulate
  return __builtin_amdgcn_wmma_f32_16x16x32_bf16(false, a, false, b, (short)0, c,
                                                 false, false);
}

__device__ __forceinline__ v16bf frag_from(const bf16* lo_p, const bf16* hi_p) {
  bf16x8 lo = *(const bf16x8*)lo_p;
  bf16x8 hi = *(const bf16x8*)hi_p;
  v16bf r;
#pragma unroll
  for (int i = 0; i < 8; ++i) { r[i] = lo[i]; r[i + 8] = hi[i]; }
  return r;
}
// A-frag 16x32: lane&15 = row; per-lane k runs {8h..8h+7, 16+8h..16+8h+7}
__device__ __forceinline__ v16bf load_frag_A(const bf16* row, int half) {
  return frag_from(row + 8 * half, row + 16 + 8 * half);
}
// B-frag 32x16: lane&15 = col; per-lane k runs {8h..8h+7, 16+8h..16+8h+7}
__device__ __forceinline__ v16bf load_frag_B(const bf16* col, int half) {
  return frag_from(col + 8 * half, col + 16 + 8 * half);
}

// ------------------------------------------------- CDNA5 async global->LDS copy
// GVS mode: per-lane LDS byte address (VDST), per-lane 32-bit byte offset
// (VADDR), wave-uniform 64-bit base (SADDR). Tracked with ASYNCcnt.
__device__ __forceinline__ void async_b128_to_lds(unsigned lds_addr,
                                                  unsigned voff,
                                                  const void* sbase) {
  asm volatile("global_load_async_to_lds_b128 %0, %1, %2"
               :: "v"(lds_addr), "v"(voff), "s"(sbase)
               : "memory");
}
__device__ __forceinline__ void wait_asynccnt0() {
  asm volatile("s_wait_asynccnt 0x0" ::: "memory");
}

// ================================================================ Kernel 1
// qkv = hidden @ w_qkv (bf16 WMMA, f32 acc), double-buffered LDS,
// fused GPT-J RoPE epilogue. Q,K,V out as bf16 (B, NH, S, HD).
#define LDSTR 48   // LDS row stride in bf16 (16B aligned, 96 B per row)

__global__ __launch_bounds__(256)
void qkv_rope_gemm(const float* __restrict__ hs, const float* __restrict__ wqkv,
                   const int* __restrict__ pos_ids,
                   bf16* __restrict__ Qo, bf16* __restrict__ Ko, bf16* __restrict__ Vo)
{
  __shared__ bf16 Abuf[2][128 * LDSTR];   // [row][k]  k-contiguous
  __shared__ bf16 Bbuf[2][128 * LDSTR];   // [n][k]    k-contiguous (transposed)

  const int tid  = threadIdx.x;
  const int wave = tid >> 5, lane = tid & 31;
  const int half = lane >> 4, l16 = lane & 15;
  const int wm = wave >> 1, wn = wave & 1;          // 4x2 wave grid
  const int m0 = blockIdx.x * 128;
  const int n0 = blockIdx.y * 128;

  // per-thread cooperative-load coordinates
  const int arow[4] = { (tid) >> 3, (tid + 256) >> 3, (tid + 512) >> 3, (tid + 768) >> 3 };
  const int ac4     = tid & 7;
  const int bkr[4]  = { (tid) >> 5, (tid + 256) >> 5, (tid + 512) >> 5, (tid + 768) >> 5 };
  const int bc4     = tid & 31;

  v8f acc[2][4];
#pragma unroll
  for (int i = 0; i < 2; ++i)
#pragma unroll
    for (int j = 0; j < 4; ++j) acc[i][j] = (v8f)(0.0f);

  // prologue: stage k0 = 0 tile into buffer 0
#pragma unroll
  for (int i = 0; i < 4; ++i) {
    float4 v = *(const float4*)(hs + (size_t)(m0 + arow[i]) * H_ + ac4 * 4);
    bf16* d = Abuf[0] + arow[i] * LDSTR + ac4 * 4;
    d[0] = (bf16)v.x; d[1] = (bf16)v.y; d[2] = (bf16)v.z; d[3] = (bf16)v.w;
  }
#pragma unroll
  for (int i = 0; i < 4; ++i) {
    float4 v = *(const float4*)(wqkv + (size_t)bkr[i] * N3_ + n0 + bc4 * 4);
    int nb = bc4 * 4;
    Bbuf[0][(nb + 0) * LDSTR + bkr[i]] = (bf16)v.x;
    Bbuf[0][(nb + 1) * LDSTR + bkr[i]] = (bf16)v.y;
    Bbuf[0][(nb + 2) * LDSTR + bkr[i]] = (bf16)v.z;
    Bbuf[0][(nb + 3) * LDSTR + bkr[i]] = (bf16)v.w;
  }

  int p = 0;
  for (int k0 = 0; k0 < H_; k0 += 32) {
    __syncthreads();
    const bool has_next = (k0 + 32) < H_;
    float4 ra[4], rb[4];
    if (has_next) {
#pragma unroll
      for (int i = 0; i < 4; ++i)
        ra[i] = *(const float4*)(hs + (size_t)(m0 + arow[i]) * H_ + (k0 + 32) + ac4 * 4);
#pragma unroll
      for (int i = 0; i < 4; ++i)
        rb[i] = *(const float4*)(wqkv + (size_t)(k0 + 32 + bkr[i]) * N3_ + n0 + bc4 * 4);
      if (k0 + 64 < H_) {   // warm GL2 two steps ahead (global_prefetch_b8)
        __builtin_prefetch(hs + (size_t)(m0 + arow[0]) * H_ + (k0 + 64) + ac4 * 4, 0, 3);
        __builtin_prefetch(wqkv + (size_t)(k0 + 64 + bkr[0]) * N3_ + n0 + bc4 * 4, 0, 3);
      }
    }

    // compute on buffer p
    {
      v16bf a[2], b[4];
#pragma unroll
      for (int mi = 0; mi < 2; ++mi)
        a[mi] = load_frag_A(Abuf[p] + (wm * 32 + mi * 16 + l16) * LDSTR, half);
#pragma unroll
      for (int ni = 0; ni < 4; ++ni)
        b[ni] = load_frag_B(Bbuf[p] + (wn * 64 + ni * 16 + l16) * LDSTR, half);
#pragma unroll
      for (int mi = 0; mi < 2; ++mi)
#pragma unroll
        for (int ni = 0; ni < 4; ++ni)
          acc[mi][ni] = wmma_bf16(a[mi], b[ni], acc[mi][ni]);
    }

    // stash next tile into the other buffer
    if (has_next) {
      const int q = p ^ 1;
#pragma unroll
      for (int i = 0; i < 4; ++i) {
        bf16* d = Abuf[q] + arow[i] * LDSTR + ac4 * 4;
        d[0] = (bf16)ra[i].x; d[1] = (bf16)ra[i].y;
        d[2] = (bf16)ra[i].z; d[3] = (bf16)ra[i].w;
      }
#pragma unroll
      for (int i = 0; i < 4; ++i) {
        int nb = bc4 * 4;
        Bbuf[q][(nb + 0) * LDSTR + bkr[i]] = (bf16)rb[i].x;
        Bbuf[q][(nb + 1) * LDSTR + bkr[i]] = (bf16)rb[i].y;
        Bbuf[q][(nb + 2) * LDSTR + bkr[i]] = (bf16)rb[i].z;
        Bbuf[q][(nb + 3) * LDSTR + bkr[i]] = (bf16)rb[i].w;
      }
    }
    p ^= 1;
  }

  // Epilogue: RoPE (GPT-J interleaved, first RD dims of Q/K heads) + bf16 store
#pragma unroll
  for (int mi = 0; mi < 2; ++mi) {
#pragma unroll
    for (int ni = 0; ni < 4; ++ni) {
      int ncol = n0 + wn * 64 + ni * 16 + l16;
      int section = ncol >> 12;         // 0:Q 1:K 2:V
      int nn = ncol & (H_ - 1);
      int h = nn >> 8;
      int dd = nn & (HD_ - 1);
      bf16* dst = (section == 0) ? Qo : (section == 1) ? Ko : Vo;
      bool do_rope = (section < 2) && (dd < RD_);   // uniform per 16-wide tile
      float inv_freq = __expf(-NEG_LOG_THETA_OVER_RD * (float)(dd & ~1));
#pragma unroll
      for (int r = 0; r < 8; ++r) {
        int m = m0 + wm * 32 + mi * 16 + r + 8 * half;
        int bidx = m >> 11, s = m & (S_ - 1);
        float val = acc[mi][ni][r];
        float partner = __shfl_xor(val, 1, 32);     // wave-uniform shuffle
        if (do_rope) {
          float ang = (float)pos_ids[m] * inv_freq;
          float c = __cosf(ang), sn = __sinf(ang);
          val = (dd & 1) ? (val * c + partner * sn) : (val * c - partner * sn);
        }
        dst[(((size_t)bidx * NH_ + h) * S_ + s) * HD_ + dd] = (bf16)val;
      }
    }
  }
}

// ================================================================ Kernel 2
// Causal flash attention. Block = 4 waves x 16 query rows; 32-key chunks.
// K chunk streamed with GLOBAL_LOAD_ASYNC_TO_LDS_B128 (ASYNCcnt path).
#define CK 32            // key-chunk length
#define KSTR 264         // Kbuf stride (256+8), bf16
#define VSTR 40          // Vt / Pbuf stride (32+8), bf16

__global__ __launch_bounds__(128)
void flash_attn(const bf16* __restrict__ Qg, const bf16* __restrict__ Kg,
                const bf16* __restrict__ Vg, bf16* __restrict__ ctx)
{
  __shared__ bf16 Kbuf[CK * KSTR];        // [s_k][d]
  __shared__ bf16 Vt[HD_ * VSTR];         // [d][s_k] (transposed)
  __shared__ bf16 Pbuf[4 * 16 * VSTR];    // per-wave P staging [row][k]

  const int tid  = threadIdx.x;
  const int wave = tid >> 5, lane = tid & 31;
  const int half = lane >> 4, l16 = lane & 15;
  const int qblk = blockIdx.x;
  const int bh   = blockIdx.y;            // b*NH + h
  const int q_base = qblk * 64 + wave * 16;
  const size_t base = (size_t)bh * S_ * HD_;

  // Q fragments resident in registers (lane reads its own row; one-time load)
  const bf16* qrow = Qg + base + (size_t)(q_base + l16) * HD_;
  v16bf qfrag[8];
#pragma unroll
  for (int dk = 0; dk < 8; ++dk) qfrag[dk] = load_frag_A(qrow + dk * 32, half);

  v8f acc_o[16];
#pragma unroll
  for (int i = 0; i < 16; ++i) acc_o[i] = (v8f)(0.0f);
  float m_r[8], l_r[8];
#pragma unroll
  for (int r = 0; r < 8; ++r) { m_r[r] = -3.0e38f; l_r[r] = 0.0f; }

  const int kc_end = 2 * qblk + 1;        // causal: chunks fully past block skipped
  for (int kc = 0; kc <= kc_end; ++kc) {
    __syncthreads();
    // --- K chunk: async global->LDS (no VGPR round trip) -------------------
    const bf16* kg_base = Kg + base + (size_t)(kc * CK) * HD_;
#pragma unroll
    for (int i = 0; i < 8; ++i) {
      int f = tid + i * 128;              // 1024 8-bf16 slots, contiguous: f*16B
      unsigned lds_a =
          (unsigned)(uintptr_t)&Kbuf[(f >> 5) * KSTR + (f & 31) * 8];
      async_b128_to_lds(lds_a, (unsigned)(f * 16), kg_base);
    }
    // --- V chunk: synchronous load + transpose into Vt ---------------------
#pragma unroll
    for (int i = 0; i < 8; ++i) {
      int f = tid + i * 128;
      int row = f >> 5, c8 = f & 31;
      bf16x8 vv = *(const bf16x8*)(Vg + base + (size_t)(kc * CK + row) * HD_ + c8 * 8);
#pragma unroll
      for (int j = 0; j < 8; ++j) Vt[(c8 * 8 + j) * VSTR + row] = vv[j];
    }
    wait_asynccnt0();
    __syncthreads();

    // scores = Q @ K^T  (B-frags come straight from row-major Kbuf)
    v8f sacc[2];
    sacc[0] = (v8f)(0.0f); sacc[1] = (v8f)(0.0f);
#pragma unroll
    for (int ni = 0; ni < 2; ++ni) {
      const bf16* kp = Kbuf + (ni * 16 + l16) * KSTR;
#pragma unroll
      for (int dk = 0; dk < 8; ++dk)
        sacc[ni] = wmma_bf16(qfrag[dk], load_frag_B(kp + dk * 32, half), sacc[ni]);
    }

    // scale + causal mask + online softmax
    float mnew[8], alpha[8];
#pragma unroll
    for (int r = 0; r < 8; ++r) {
      int sq = q_base + r + 8 * half;
      float mx = m_r[r];
#pragma unroll
      for (int ni = 0; ni < 2; ++ni) {
        float v = sacc[ni][r] * SCALE_;
        int sk = kc * CK + ni * 16 + l16;
        if (sk > sq) v = -3.0e38f;
        sacc[ni][r] = v;
        mx = fmaxf(mx, v);
      }
      mx = fmaxf(mx, __shfl_xor(mx, 1, 32));
      mx = fmaxf(mx, __shfl_xor(mx, 2, 32));
      mx = fmaxf(mx, __shfl_xor(mx, 4, 32));
      mx = fmaxf(mx, __shfl_xor(mx, 8, 32));
      mnew[r] = mx;
    }
#pragma unroll
    for (int r = 0; r < 8; ++r) {
      float s = 0.0f;
#pragma unroll
      for (int ni = 0; ni < 2; ++ni) {
        float pv = __expf(sacc[ni][r] - mnew[r]);
        sacc[ni][r] = pv;
        s += pv;
      }
      s += __shfl_xor(s, 1, 32);
      s += __shfl_xor(s, 2, 32);
      s += __shfl_xor(s, 4, 32);
      s += __shfl_xor(s, 8, 32);
      alpha[r] = __expf(m_r[r] - mnew[r]);
      l_r[r] = l_r[r] * alpha[r] + s;
      m_r[r] = mnew[r];
    }
#pragma unroll
    for (int dt = 0; dt < 16; ++dt)
#pragma unroll
      for (int r = 0; r < 8; ++r) acc_o[dt][r] *= alpha[r];

    // P: C-layout -> A-layout through per-wave LDS (same-wave, no barrier)
    bf16* pb = Pbuf + wave * 16 * VSTR;
#pragma unroll
    for (int ni = 0; ni < 2; ++ni)
#pragma unroll
      for (int r = 0; r < 8; ++r)
        pb[(r + 8 * half) * VSTR + ni * 16 + l16] = (bf16)sacc[ni][r];
    v16bf pf = load_frag_A(pb + l16 * VSTR, half);

    // ctx += P @ V   (B-frags from transposed Vt: k-contiguous)
#pragma unroll
    for (int dt = 0; dt < 16; ++dt)
      acc_o[dt] = wmma_bf16(pf, load_frag_B(Vt + (dt * 16 + l16) * VSTR, half),
                            acc_o[dt]);
  }

  // epilogue: normalize, store bf16 ctx as (B*S, H) rows for output GEMM
  const int bidx = bh >> 4, h = bh & (NH_ - 1);
#pragma unroll
  for (int dt = 0; dt < 16; ++dt)
#pragma unroll
    for (int r = 0; r < 8; ++r) {
      int sq = q_base + r + 8 * half;
      size_t o = ((size_t)bidx * S_ + sq) * H_ + h * HD_ + dt * 16 + l16;
      ctx[o] = (bf16)(acc_o[dt][r] / l_r[r]);
    }
}

// ================================================================ Kernel 3
// out = ctx(bf16) @ w_out(f32->bf16), f32 result. Double-buffered LDS.
__global__ __launch_bounds__(256)
void out_proj_gemm(const bf16* __restrict__ ctx, const float* __restrict__ wout,
                   float* __restrict__ out)
{
  __shared__ bf16 Abuf[2][128 * LDSTR];
  __shared__ bf16 Bbuf[2][128 * LDSTR];

  const int tid  = threadIdx.x;
  const int wave = tid >> 5, lane = tid & 31;
  const int half = lane >> 4, l16 = lane & 15;
  const int wm = wave >> 1, wn = wave & 1;
  const int m0 = blockIdx.x * 128;
  const int n0 = blockIdx.y * 128;

  const int arow[2] = { tid >> 2, (tid + 256) >> 2 };    // 512 8-bf16 slots
  const int ac8     = tid & 3;
  const int bkr[4]  = { tid >> 5, (tid + 256) >> 5, (tid + 512) >> 5, (tid + 768) >> 5 };
  const int bc4     = tid & 31;

  v8f acc[2][4];
#pragma unroll
  for (int i = 0; i < 2; ++i)
#pragma unroll
    for (int j = 0; j < 4; ++j) acc[i][j] = (v8f)(0.0f);

  // prologue: stage k0 = 0 tile
#pragma unroll
  for (int i = 0; i < 2; ++i)
    *(bf16x8*)(Abuf[0] + arow[i] * LDSTR + ac8 * 8) =
        *(const bf16x8*)(ctx + (size_t)(m0 + arow[i]) * H_ + ac8 * 8);
#pragma unroll
  for (int i = 0; i < 4; ++i) {
    float4 v = *(const float4*)(wout + (size_t)bkr[i] * H_ + n0 + bc4 * 4);
    int nb = bc4 * 4;
    Bbuf[0][(nb + 0) * LDSTR + bkr[i]] = (bf16)v.x;
    Bbuf[0][(nb + 1) * LDSTR + bkr[i]] = (bf16)v.y;
    Bbuf[0][(nb + 2) * LDSTR + bkr[i]] = (bf16)v.z;
    Bbuf[0][(nb + 3) * LDSTR + bkr[i]] = (bf16)v.w;
  }

  int p = 0;
  for (int k0 = 0; k0 < H_; k0 += 32) {
    __syncthreads();
    const bool has_next = (k0 + 32) < H_;
    bf16x8 ra[2];
    float4 rb[4];
    if (has_next) {
#pragma unroll
      for (int i = 0; i < 2; ++i)
        ra[i] = *(const bf16x8*)(ctx + (size_t)(m0 + arow[i]) * H_ + (k0 + 32) + ac8 * 8);
#pragma unroll
      for (int i = 0; i < 4; ++i)
        rb[i] = *(const float4*)(wout + (size_t)(k0 + 32 + bkr[i]) * H_ + n0 + bc4 * 4);
      if (k0 + 64 < H_) {
        __builtin_prefetch(ctx + (size_t)(m0 + arow[0]) * H_ + (k0 + 64) + ac8 * 8, 0, 3);
        __builtin_prefetch(wout + (size_t)(k0 + 64 + bkr[0]) * H_ + n0 + bc4 * 4, 0, 3);
      }
    }

    {
      v16bf a[2], b[4];
#pragma unroll
      for (int mi = 0; mi < 2; ++mi)
        a[mi] = load_frag_A(Abuf[p] + (wm * 32 + mi * 16 + l16) * LDSTR, half);
#pragma unroll
      for (int ni = 0; ni < 4; ++ni)
        b[ni] = load_frag_B(Bbuf[p] + (wn * 64 + ni * 16 + l16) * LDSTR, half);
#pragma unroll
      for (int mi = 0; mi < 2; ++mi)
#pragma unroll
        for (int ni = 0; ni < 4; ++ni)
          acc[mi][ni] = wmma_bf16(a[mi], b[ni], acc[mi][ni]);
    }

    if (has_next) {
      const int q = p ^ 1;
#pragma unroll
      for (int i = 0; i < 2; ++i)
        *(bf16x8*)(Abuf[q] + arow[i] * LDSTR + ac8 * 8) = ra[i];
#pragma unroll
      for (int i = 0; i < 4; ++i) {
        int nb = bc4 * 4;
        Bbuf[q][(nb + 0) * LDSTR + bkr[i]] = (bf16)rb[i].x;
        Bbuf[q][(nb + 1) * LDSTR + bkr[i]] = (bf16)rb[i].y;
        Bbuf[q][(nb + 2) * LDSTR + bkr[i]] = (bf16)rb[i].z;
        Bbuf[q][(nb + 3) * LDSTR + bkr[i]] = (bf16)rb[i].w;
      }
    }
    p ^= 1;
  }

#pragma unroll
  for (int mi = 0; mi < 2; ++mi)
#pragma unroll
    for (int ni = 0; ni < 4; ++ni) {
      int ncol = n0 + wn * 64 + ni * 16 + l16;
#pragma unroll
      for (int r = 0; r < 8; ++r) {
        int m = m0 + wm * 32 + mi * 16 + r + 8 * half;
        out[(size_t)m * H_ + ncol] = acc[mi][ni][r];
      }
    }
}

// ================================================================ launch
extern "C" void kernel_launch(void* const* d_in, const int* in_sizes, int n_in,
                              void* d_out, int out_size, void* d_ws, size_t ws_size,
                              hipStream_t stream) {
  (void)in_sizes; (void)n_in; (void)out_size; (void)ws_size;
  const int*   pos  = (const int*)d_in[0];
  const float* hs   = (const float*)d_in[1];
  const float* wqkv = (const float*)d_in[2];
  const float* wout = (const float*)d_in[3];
  float* out = (float*)d_out;

  const size_t NE = (size_t)B_ * NH_ * S_ * HD_;   // 33,554,432 elems per tensor
  bf16* Qw = (bf16*)d_ws;
  bf16* Kw = Qw + NE;
  bf16* Vw = Kw + NE;
  bf16* Cw = Vw + NE;

  dim3 gA(8192 / 128, N3_ / 128);                  // 64 x 96
  qkv_rope_gemm<<<gA, 256, 0, stream>>>(hs, wqkv, pos, Qw, Kw, Vw);

  dim3 gF(S_ / 64, B_ * NH_);                      // 32 x 64
  flash_attn<<<gF, 128, 0, stream>>>(Qw, Kw, Vw, Cw);

  dim3 gC(8192 / 128, H_ / 128);                   // 64 x 32
  out_proj_gemm<<<gC, 256, 0, stream>>>(Cw, wout, out);
}
